// PairwiseFusion_6554120094053
// MI455X (gfx1250) — matrix operand order repaired and checked by reading, not compiled, not run on hardware
//
#include <hip/hip_runtime.h>
#include <hip/hip_bf16.h>
#include <math.h>

// ---------------------------------------------------------------------------
// PairwiseFusion for MI455X (gfx1250, wave32, WMMA + TDM + async-LDS).
//
// Roofline: 722 GFLOP vs <1 GB traffic @ 23.3 TB/s -> compute bound; all three
// GEMMs run v_wmma_f32_16x16x32_bf16 (K=32/issue vs K=4 for f32 WMMA), f32
// accumulate, f32 epilogues (bias/GELU/LayerNorm).
//
// Data movement (CDNA5-specific):
//  * A (activation) tiles: per-lane global_load_async_to_lds_b128 (ASYNCcnt)
//  * B (weight) tiles:     one tensor_load_to_lds TDM descriptor per K-step
//                          (2D tile 32x128 elems, stride=ldb), TENSORcnt
//  * double-buffered LDS (2 x 8KB per matrix), one barrier per K-step:
//      wait(t) -> barrier -> issue(t+1) -> 8x WMMA on tile t
// ---------------------------------------------------------------------------

typedef __attribute__((ext_vector_type(16))) __bf16 bf16x16;
typedef __attribute__((ext_vector_type(8)))  __bf16 bf16x8;
typedef __attribute__((ext_vector_type(8)))  float  f32x8;
typedef __attribute__((ext_vector_type(4)))  unsigned int u32x4;
typedef __attribute__((ext_vector_type(8)))  unsigned int u32x8;

#define N_ROWS  16384
#define D_DIM   1024
#define HID_DIM 4096
#define INDIM   4097
#define K1PAD   4128   // 4097 rounded to multiple of 32 (zero padded)

#define BM 128
#define BN 128
#define BK 32
#define TPB 256        // 8 wave32 waves: 2 wave-rows x 4 wave-cols
#define TILE_BYTES (BM * BK * 2)   // 8 KB per buffer

__device__ __forceinline__ float gelu_exact(float x) {
    return 0.5f * x * (1.0f + erff(x * 0.70710678118654752f));
}

// --- CDNA5 async copy: 32 contiguous bytes global -> LDS (2x B128) ----------
// VDST = per-lane LDS byte address, VADDR = per-lane 64-bit global address;
// the instruction offset applies to both sides (ISA 10.x async pseudocode).
__device__ __forceinline__ void async_copy_32B(unsigned lds_off, const __bf16* g) {
    asm volatile("global_load_async_to_lds_b128 %0, %1, off\n\t"
                 "global_load_async_to_lds_b128 %0, %1, off offset:16"
                 :: "v"(lds_off), "v"(g)
                 : "memory");
}

// --- CDNA5 TDM: 2D tile (tile_dim0=32 elems x tile_dim1=128 rows, 2B elems)
// from global (row stride = ld elems) into contiguous LDS.  D# per ISA 8.3/8.4:
//   group0: count=1 | lds_addr | global_addr[56:0] | type=2
//   group1: data_size=1(2B), tensor_dim0=ld, tensor_dim1=2^20, tile 32x128,
//           tensor_dim0_stride=ld, everything else 0 (no pad/iterate/multicast)
__device__ __forceinline__ void tdm_load_tile(unsigned lds_off, const __bf16* gtile,
                                              unsigned ld) {
    const unsigned long long ga = (unsigned long long)(size_t)gtile;
    u32x4 g0;
    g0[0] = 1u;                                            // count=1, user D#
    g0[1] = lds_off;                                       // lds_addr
    g0[2] = (unsigned)ga;                                  // global_addr[31:0]
    g0[3] = (unsigned)((ga >> 32) & 0x01ffffffu) | (2u << 30); // addr[56:32]|type=2
    const unsigned dim1 = 1u << 20;                        // generous tensor_dim1
    u32x8 g1;
    g1[0] = 1u << 16;                                      // data_size=1 -> 2B
    g1[1] = (ld & 0xffffu) << 16;                          // tensor_dim0[15:0]
    g1[2] = (ld >> 16) | ((dim1 & 0xffffu) << 16);         // dim0[31:16]|dim1[15:0]
    g1[3] = (dim1 >> 16) | (32u << 16);                    // dim1[31:16]|tile_dim0=32
    g1[4] = 128u;                                          // tile_dim1=128, tile_dim2=0
    g1[5] = ld;                                            // tensor_dim0_stride[31:0]
    g1[6] = 0u;                                            // stride[47:32]=0, d1s lo=0
    g1[7] = 0u;                                            // tensor_dim1_stride hi=0
    asm volatile("tensor_load_to_lds %0, %1" :: "s"(g0), "s"(g1) : "memory");
}

// --- WMMA fragment loads from LDS tiles (ISA 7.12.2 VGPR layouts) -----------
__device__ __forceinline__ bf16x16 load_a_frag(const __bf16* As, int rowBase, int lane) {
    const int r  = rowBase + (lane & 15);
    const int kb = (lane >> 4) * 8;
    bf16x8 lo = *(const bf16x8*)(As + r * BK + kb);
    bf16x8 hi = *(const bf16x8*)(As + r * BK + kb + 16);
    return __builtin_shufflevector(lo, hi, 0,1,2,3,4,5,6,7,8,9,10,11,12,13,14,15);
}
__device__ __forceinline__ bf16x16 load_b_frag(const __bf16* Bs, int colBase, int lane) {
    const int c  = colBase + (lane & 15);
    const int kb = (lane >> 4) * 8;
    bf16x8 lo = *(const bf16x8*)(Bs + c * BK + kb);
    bf16x8 hi = *(const bf16x8*)(Bs + c * BK + kb + 16);
    return __builtin_shufflevector(lo, hi, 0,1,2,3,4,5,6,7,8,9,10,11,12,13,14,15);
}

// --- shared double-buffered WMMA mainloop -----------------------------------
__device__ __forceinline__ void gemm_core(const __bf16* __restrict__ A, int lda,
                                          const __bf16* __restrict__ Bt, int ldb,
                                          int K, int row0, int col0,
                                          __bf16* As2, __bf16* Bs2,  // [2][tile]
                                          f32x8 acc[4][2]) {
    const int tid  = threadIdx.x;
    const int lane = tid & 31;
    const int wave = tid >> 5;
    const int wm   = (wave >> 2) * 64;
    const int wn   = (wave & 3) * 32;
    const int ldr  = tid >> 1;            // tile row this thread stages
    const int ldh  = (tid & 1) * 16;      // 16-elem half of the 32-elem row

    const unsigned asA = (unsigned)(size_t)As2;   // LDS byte offsets
    const unsigned asB = (unsigned)(size_t)Bs2;
    const unsigned ldsA = asA + (unsigned)(ldr * BK + ldh) * 2u;

    const int T = K / BK;

    // prologue: tile 0 into buffer 0
    async_copy_32B(ldsA, A + (size_t)(row0 + ldr) * lda + ldh);
    if (wave == 0)
        tdm_load_tile(asB, Bt + (size_t)col0 * ldb, (unsigned)ldb);

    for (int t = 0; t < T; ++t) {
        const int buf = t & 1;
        asm volatile("s_wait_asynccnt 0x0" ::: "memory");  // own A-tile chunks landed
        __builtin_amdgcn_s_wait_tensorcnt(0);              // wave0's TDM landed
        __syncthreads();                                   // tile t visible to all

        if (t + 1 < T) {                                   // stage tile t+1
            const int nb = (t + 1) & 1;
            const int k1 = (t + 1) * BK;
            async_copy_32B(ldsA + (unsigned)(nb * TILE_BYTES),
                           A + (size_t)(row0 + ldr) * lda + k1 + ldh);
            if (wave == 0)
                tdm_load_tile(asB + (unsigned)(nb * TILE_BYTES),
                              Bt + (size_t)col0 * ldb + k1, (unsigned)ldb);
        }

        const __bf16* Ab = As2 + buf * (BM * BK);
        const __bf16* Bb = Bs2 + buf * (BN * BK);
        bf16x16 af[4], bfr[2];
#pragma unroll
        for (int f = 0; f < 4; ++f) af[f] = load_a_frag(Ab, wm + f * 16, lane);
#pragma unroll
        for (int g = 0; g < 2; ++g) bfr[g] = load_b_frag(Bb, wn + g * 16, lane);
#pragma unroll
        for (int f = 0; f < 4; ++f)
#pragma unroll
            for (int g = 0; g < 2; ++g)
                acc[f][g] = __builtin_amdgcn_wmma_f32_16x16x32_bf16(
                    false, af[f], false, bfr[g], (short)0, acc[f][g], false, false);
    }
}

// C = GELU(A[M][K] * Bt[N][K]^T + bias), out bf16 or f32.
template <bool OUT_BF16>
__global__ __launch_bounds__(TPB)
void gemm_bias_gelu(const __bf16* __restrict__ A, int lda,
                    const __bf16* __restrict__ Bt, int ldb,
                    const float*  __restrict__ bias,
                    void* __restrict__ Cout, int ldc, int K) {
    __shared__ __bf16 As2[2 * BM * BK];
    __shared__ __bf16 Bs2[2 * BN * BK];

    const int tid  = threadIdx.x;
    const int lane = tid & 31;
    const int wave = tid >> 5;
    const int wm   = (wave >> 2) * 64;
    const int wn   = (wave & 3) * 32;
    const int row0 = blockIdx.x * BM;
    const int col0 = blockIdx.y * BN;

    f32x8 acc[4][2];
#pragma unroll
    for (int f = 0; f < 4; ++f)
#pragma unroll
        for (int g = 0; g < 2; ++g)
#pragma unroll
            for (int i = 0; i < 8; ++i) acc[f][g][i] = 0.0f;

    gemm_core(A, lda, Bt, ldb, K, row0, col0, As2, Bs2, acc);

    const int mlo = (lane >> 4) * 8;
    const int nn  = lane & 15;
#pragma unroll
    for (int f = 0; f < 4; ++f) {
#pragma unroll
        for (int g = 0; g < 2; ++g) {
            const int col = col0 + wn + g * 16 + nn;
            const float bv = bias[col];
#pragma unroll
            for (int v = 0; v < 8; ++v) {
                const int row = row0 + wm + f * 16 + mlo + v;
                const float val = gelu_exact(acc[f][g][v] + bv);
                if (OUT_BF16)
                    ((__bf16*)Cout)[(size_t)row * ldc + col] = (__bf16)val;
                else
                    ((float*)Cout)[(size_t)row * ldc + col] = val;
            }
        }
    }
}

// s[i] += sum_j (zA @ W)[i,j] * zB[i,j]  (bilinear GEMM fused with row dot)
__global__ __launch_bounds__(TPB)
void bilinear_gemm_dot(const __bf16* __restrict__ A,    // zA bf16 [N][D]
                       const __bf16* __restrict__ Bt,   // bilinear^T bf16 [D][D]
                       const float*  __restrict__ zB,   // f32 [N][D]
                       float* __restrict__ s) {
    __shared__ __bf16 As2[2 * BM * BK];
    __shared__ __bf16 Bs2[2 * BN * BK];
    __shared__ float  sp[BM];

    const int tid  = threadIdx.x;
    const int lane = tid & 31;
    const int wave = tid >> 5;
    const int wm   = (wave >> 2) * 64;
    const int wn   = (wave & 3) * 32;
    const int row0 = blockIdx.x * BM;
    const int col0 = blockIdx.y * BN;

    if (tid < BM) sp[tid] = 0.0f;

    f32x8 acc[4][2];
#pragma unroll
    for (int f = 0; f < 4; ++f)
#pragma unroll
        for (int g = 0; g < 2; ++g)
#pragma unroll
            for (int i = 0; i < 8; ++i) acc[f][g][i] = 0.0f;

    gemm_core(A, D_DIM, Bt, D_DIM, D_DIM, row0, col0, As2, Bs2, acc);

    const int mlo = (lane >> 4) * 8;
    const int nn  = lane & 15;
#pragma unroll
    for (int f = 0; f < 4; ++f) {
#pragma unroll
        for (int v = 0; v < 8; ++v) {
            const int mloc = wm + f * 16 + mlo + v;
            const int row  = row0 + mloc;
            float p = 0.0f;
#pragma unroll
            for (int g = 0; g < 2; ++g) {
                const int col = col0 + wn + g * 16 + nn;
                p += acc[f][g][v] * zB[(size_t)row * D_DIM + col];
            }
            atomicAdd(&sp[mloc], p);            // ds_add_f32
        }
    }
    __syncthreads();
    if (tid < BM) atomicAdd(&s[row0 + tid], sp[tid]);   // global fp32 atomic
}

// Build x = LN(concat[zA,zB,zA*zB,|zA-zB|,s]) -> bf16 padded to K1PAD.
__global__ __launch_bounds__(256)
void build_x_ln(const float* __restrict__ zA, const float* __restrict__ zB,
                const float* __restrict__ s,
                const float* __restrict__ ln_g, const float* __restrict__ ln_b,
                __bf16* __restrict__ xout) {
    const int row = blockIdx.x;
    const int tid = threadIdx.x;
    const float* a = zA + (size_t)row * D_DIM;
    const float* b = zB + (size_t)row * D_DIM;

    float sum = 0.0f, sq = 0.0f;
    for (int j = tid; j < D_DIM; j += 256) {
        const float av = a[j], bv = b[j];
        const float hv = av * bv, dv = fabsf(av - bv);
        sum += av + bv + hv + dv;
        sq  += av * av + bv * bv + hv * hv + dv * dv;
    }
    if (tid == 0) { const float sv = s[row]; sum += sv; sq += sv * sv; }

    __shared__ float rs[256], rq[256];
    rs[tid] = sum; rq[tid] = sq;
    __syncthreads();
    for (int off = 128; off > 0; off >>= 1) {
        if (tid < off) { rs[tid] += rs[tid + off]; rq[tid] += rq[tid + off]; }
        __syncthreads();
    }
    const float mu   = rs[0] * (1.0f / (float)INDIM);
    const float var  = rq[0] * (1.0f / (float)INDIM) - mu * mu;
    const float rstd = rsqrtf(var + 1e-5f);

    __bf16* xr = xout + (size_t)row * K1PAD;
    for (int j = tid; j < D_DIM; j += 256) {
        const float av = a[j], bv = b[j];
        const float hv = av * bv, dv = fabsf(av - bv);
        xr[j]             = (__bf16)(((av - mu) * rstd) * ln_g[j]             + ln_b[j]);
        xr[D_DIM + j]     = (__bf16)(((bv - mu) * rstd) * ln_g[D_DIM + j]     + ln_b[D_DIM + j]);
        xr[2 * D_DIM + j] = (__bf16)(((hv - mu) * rstd) * ln_g[2 * D_DIM + j] + ln_b[2 * D_DIM + j]);
        xr[3 * D_DIM + j] = (__bf16)(((dv - mu) * rstd) * ln_g[3 * D_DIM + j] + ln_b[3 * D_DIM + j]);
    }
    if (tid == 0)
        xr[4096] = (__bf16)(((s[row] - mu) * rstd) * ln_g[4096] + ln_b[4096]);
    if (tid >= 1 && tid < 32) xr[4096 + tid] = (__bf16)0.0f;   // zero pad
}

// --- small utility kernels ---------------------------------------------------
__global__ void f32_to_bf16(const float* __restrict__ in, __bf16* __restrict__ out, int n) {
    int i = blockIdx.x * blockDim.x + threadIdx.x;
    const int stride = gridDim.x * blockDim.x;
    for (; i < n; i += stride) out[i] = (__bf16)in[i];
}

__global__ void zero_f32(float* __restrict__ p, int n) {
    int i = blockIdx.x * blockDim.x + threadIdx.x;
    const int stride = gridDim.x * blockDim.x;
    for (; i < n; i += stride) p[i] = 0.0f;
}

// in[K][Nn] f32 (row-major) -> out[Nn][Kpad] bf16 (K contiguous, zero padded).
__global__ void transpose_pad_bf16(const float* __restrict__ in, __bf16* __restrict__ out,
                                   int K, int Nn, int Kpad) {
    __shared__ float t[32][33];
    const int kbase = blockIdx.x * 32;
    const int nbase = blockIdx.y * 32;
    for (int i = threadIdx.y; i < 32; i += 8) {
        const int k = kbase + i;
        t[i][threadIdx.x] = (k < K) ? in[(size_t)k * Nn + nbase + threadIdx.x] : 0.0f;
    }
    __syncthreads();
    for (int i = threadIdx.y; i < 32; i += 8) {
        const int n = nbase + i;
        const int k = kbase + threadIdx.x;
        out[(size_t)n * Kpad + k] = (__bf16)t[threadIdx.x][i];
    }
}

// ---------------------------------------------------------------------------
extern "C" void kernel_launch(void* const* d_in, const int* in_sizes, int n_in,
                              void* d_out, int out_size, void* d_ws, size_t ws_size,
                              hipStream_t stream) {
    (void)in_sizes; (void)n_in; (void)out_size; (void)ws_size;
    const float* zA   = (const float*)d_in[0];
    const float* zB   = (const float*)d_in[1];
    const float* bil  = (const float*)d_in[2];
    const float* ln_g = (const float*)d_in[3];
    const float* ln_b = (const float*)d_in[4];
    const float* W1   = (const float*)d_in[5];
    const float* b1   = (const float*)d_in[6];
    const float* W2   = (const float*)d_in[7];
    const float* b2   = (const float*)d_in[8];
    float* out = (float*)d_out;

    char* w = (char*)d_ws;
    __bf16* zAb  = (__bf16*)w;  w += (size_t)N_ROWS * D_DIM * 2;
    __bf16* Wbt  = (__bf16*)w;  w += (size_t)D_DIM * D_DIM * 2;
    float*  sbuf = (float*)w;   w += (size_t)N_ROWS * 4;
    __bf16* xb   = (__bf16*)w;  w += (size_t)N_ROWS * K1PAD * 2;
    __bf16* W1t  = (__bf16*)w;  w += (size_t)HID_DIM * K1PAD * 2;
    __bf16* hb   = (__bf16*)w;  w += (size_t)N_ROWS * HID_DIM * 2;
    __bf16* W2t  = (__bf16*)w;  /* total ~348 MB */

    const dim3 tb(32, 8);
    f32_to_bf16<<<2048, 256, 0, stream>>>(zA, zAb, N_ROWS * D_DIM);
    transpose_pad_bf16<<<dim3(D_DIM / 32, D_DIM / 32), tb, 0, stream>>>(bil, Wbt, D_DIM, D_DIM, D_DIM);
    transpose_pad_bf16<<<dim3(K1PAD / 32, HID_DIM / 32), tb, 0, stream>>>(W1, W1t, INDIM, HID_DIM, K1PAD);
    transpose_pad_bf16<<<dim3(HID_DIM / 32, D_DIM / 32), tb, 0, stream>>>(W2, W2t, HID_DIM, D_DIM, HID_DIM);
    zero_f32<<<64, 256, 0, stream>>>(sbuf, N_ROWS);

    // s[i] = zA[i] @ bilinear @ zB[i]^T
    bilinear_gemm_dot<<<dim3(N_ROWS / BM, D_DIM / BN), TPB, 0, stream>>>(zAb, Wbt, zB, sbuf);

    // x = LayerNorm(concat) -> bf16 padded
    build_x_ln<<<N_ROWS, 256, 0, stream>>>(zA, zB, sbuf, ln_g, ln_b, xb);

    // h = GELU(x @ W1 + b1)  [dominant 550 GFLOP GEMM]
    gemm_bias_gelu<true><<<dim3(N_ROWS / BM, HID_DIM / BN), TPB, 0, stream>>>(
        xb, K1PAD, W1t, K1PAD, b1, hb, HID_DIM, K1PAD);

    // out = GELU(h @ W2 + b2) -> f32
    gemm_bias_gelu<false><<<dim3(N_ROWS / BM, D_DIM / BN), TPB, 0, stream>>>(
        hb, HID_DIM, W2t, HID_DIM, b2, out, D_DIM, HID_DIM);
}
